// FINN_Burger2D_20864951124293
// MI455X (gfx1250) — matrix operand order repaired and checked by reading, not compiled, not run on hardware
//
#include <hip/hip_runtime.h>
#include <math.h>

// ---------------------------------------------------------------------------
// FINN Burger2D flux kernel for MI455X (gfx1250, wave32).
//
// Strategy:
//   * The 32x32 MLP middle layer is computed as H2^T = W2^T * H1^T using
//     v_wmma_f32_16x16x32_f16:
//       A (16x32 f16) = a 16-column half of W2^T  (constant, built once/wave)
//       B (32x16 f16) = H1^T for 16 grid points   (built per tile, 16 tanh/lane)
//       D (16x16 f32) = H2^T half (hidden units x points)
//   * D layout puts hidden-unit rows in-lane => W3 contraction is 8 in-lane
//     FMAs per D matrix plus ONE __shfl_xor(16); the scalar advection coeff
//     `a` lands on exactly the lane owning that grid point.
//   * Each wave handles 32 points/iter (two 16-pt tiles, 4 WMMAs, shared A),
//     grid-stride over all 4.19M points. tanh work = 64/point (no redundancy).
//   * Stencil/BC part is pure VALU + cached global loads (neighbors hit L2).
// ---------------------------------------------------------------------------

typedef __attribute__((ext_vector_type(16))) _Float16 v16h;
typedef __attribute__((ext_vector_type(8)))  float    v8f;

#define NXg 2048
#define NYg 2048
#define INV_DX 100.0f   // 1/0.01
#define INV_DY 100.0f

#if __has_builtin(__builtin_amdgcn_tanhf)
#define TANHF(x) __builtin_amdgcn_tanhf(x)
#else
#define TANHF(x) tanhf(x)
#endif

__device__ __forceinline__ v8f wmma_f16(v16h a, v16h b, v8f c) {
  // 8 args: (neg_a, A, neg_b, B, c_mod, C, reuse_a, reuse_b)
  return __builtin_amdgcn_wmma_f32_16x16x32_f16(false, a, false, b,
                                                (short)0, c, false, false);
}

__global__ __launch_bounds__(256)
void finn_burger2d_kernel(const float* __restrict__ u,
                          const float* __restrict__ W1,   // [1,32]  flat 32
                          const float* __restrict__ W2,   // [32,32] row-major
                          const float* __restrict__ W3,   // [32,1]  flat 32
                          const float* __restrict__ Dc,   // [1]
                          const float* __restrict__ BC,   // [2]
                          const float* __restrict__ st,   // [2] stencil
                          float* __restrict__ out)
{
  const int l = threadIdx.x & 31;   // lane in wave
  const int g = l >> 4;             // half-wave group
  const int n = l & 15;             // column / point-in-tile index

  // ---- scalar params (uniform, cached) ----
  const float d   = Dc[0];
  const float bc0 = BC[0];
  const float bc1 = BC[1];
  const float s0  = st[0];
  const float s1  = st[1];

  // ---- constant A operands: A_lo/A_hi = 16-col halves of W2^T (16x32 f16) ---
  // 16-bit A layout (ISA 7.12.2): lane row M = l&15; VGPR v pair:
  //   v<4 : K = 2v + 8g, 2v+1 + 8g        v>=4: K = 16 + 2(v-4) + 8g, +1
  v16h Alo{}, Ahi{};
#pragma unroll
  for (int v = 0; v < 8; ++v) {
    const int k0 = (v < 4) ? (2 * v + 8 * g) : (16 + 2 * (v - 4) + 8 * g);
    const int k1 = k0 + 1;
    // A_lo[M=n][K=k] = W2^T[n][k] = W2[k][n]
    Alo[2 * v]     = (_Float16)W2[k0 * 32 + n];
    Alo[2 * v + 1] = (_Float16)W2[k1 * 32 + n];
    Ahi[2 * v]     = (_Float16)W2[k0 * 32 + 16 + n];
    Ahi[2 * v + 1] = (_Float16)W2[k1 * 32 + 16 + n];
  }

  // ---- per-lane weight slices kept in registers ----
  // B-building needs W1[k] for this lane's K set: k = 8g + i (i<8), 16 + 8g + (i-8) (i>=8)
  float w1r[16];
#pragma unroll
  for (int i = 0; i < 16; ++i) w1r[i] = W1[(i < 8) ? (8 * g + i) : (16 + 8 * g + (i - 8))];
  // D reg r holds hidden unit (r + 8g) for the lo-half, (16 + r + 8g) for hi.
  float w3a[8], w3b[8];
#pragma unroll
  for (int r = 0; r < 8; ++r) {
    w3a[r] = W3[8 * g + r];
    w3b[r] = W3[16 + 8 * g + r];
  }

  // ---- grid-stride loop: one iteration = 32 consecutive points ----
  const int wavesPerBlock = blockDim.x >> 5;
  const int wave   = blockIdx.x * wavesPerBlock + (threadIdx.x >> 5);
  const int nWaves = gridDim.x * wavesPerBlock;
  const int NT     = (NXg * NYg) / 32;   // 131072 tiles of 32 points

  for (int t = wave; t < NT; t += nWaves) {
    const int base = t * 32;
    const int p = base + l;           // this lane's grid point
    const int x = p >> 11;            // NY = 2048
    const int y = p & (NYg - 1);

    const float uc = u[p];
    const float uo = __shfl_xor(uc, 16);          // partner half's value
    const float u0 = (g == 0) ? uc : uo;          // point base + n    (tile 0)
    const float u1 = (g == 0) ? uo : uc;          // point base + 16+n (tile 1)

    // ---- B = H1^T (32x16 f16): lane col N = n, K = 8g + i (i<8), 16 + 8g + (i-8) (i>=8) --
    v16h B0{}, B1{};
#pragma unroll
    for (int i = 0; i < 16; ++i) {
      B0[i] = (_Float16)TANHF(u0 * w1r[i]);
      B1[i] = (_Float16)TANHF(u1 * w1r[i]);
    }

    // ---- 4 WMMAs: two output-unit halves x two point tiles -----------------
    v8f z{};
    v8f D00 = wmma_f16(Alo, B0, z);   // units  0..15, tile 0
    v8f D10 = wmma_f16(Ahi, B0, z);   // units 16..31, tile 0
    v8f D01 = wmma_f16(Alo, B1, z);   // units  0..15, tile 1
    v8f D11 = wmma_f16(Ahi, B1, z);   // units 16..31, tile 1

    // ---- output layer: a = sum_j tanh(H2[j]) * W3[j] -----------------------
    float p0 = 0.0f, p1 = 0.0f;
#pragma unroll
    for (int r = 0; r < 8; ++r) {
      p0 = fmaf(TANHF(D00[r]), w3a[r], p0);
      p0 = fmaf(TANHF(D10[r]), w3b[r], p0);
      p1 = fmaf(TANHF(D01[r]), w3a[r], p1);
      p1 = fmaf(TANHF(D11[r]), w3b[r], p1);
    }
    // lanes g=0 hold units {0-7,16-23}, g=1 holds {8-15,24-31}: one xor-16 sum
    const float a0 = p0 + __shfl_xor(p0, 16);     // a for tile-0 point n
    const float a1 = p1 + __shfl_xor(p1, 16);     // a for tile-1 point n
    const float a  = (g == 0) ? a0 : a1;          // == a for point p = base+l

    // ---- 5-point flux stencil with BC fill ---------------------------------
    const float uL = (x > 0)        ? u[p - NYg] : bc0;
    const float uR = (x < NXg - 1)  ? u[p + NYg] : bc1;
    const float uB = (y > 0)        ? u[p - 1]   : bc0;
    const float uT = (y < NYg - 1)  ? u[p + 1]   : bc1;

    const float ap = fmaxf(a, 0.0f);
    const float am = fminf(a, 0.0f);

    const float gl = s0 * uc + s1 * uL;
    const float gr = s0 * uc + s1 * uR;
    const float gb = s0 * uc + s1 * uB;
    const float gt = s0 * uc + s1 * uT;

    const float fl = d * gl + ap * INV_DX * gl;   // -ap*(-gl) = +ap*gl
    const float fr = d * gr - am * INV_DX * gr;
    const float fb = d * gb + ap * INV_DY * gb;
    const float ft = d * gt - am * INV_DY * gt;

    out[p] = fl + fr + fb + ft;
  }
}

extern "C" void kernel_launch(void* const* d_in, const int* in_sizes, int n_in,
                              void* d_out, int out_size, void* d_ws, size_t ws_size,
                              hipStream_t stream) {
  (void)in_sizes; (void)n_in; (void)out_size; (void)d_ws; (void)ws_size;
  const float* u   = (const float*)d_in[0];
  const float* W1  = (const float*)d_in[1];
  const float* W2  = (const float*)d_in[2];
  const float* W3  = (const float*)d_in[3];
  const float* Dc  = (const float*)d_in[4];
  const float* BC  = (const float*)d_in[5];
  const float* st  = (const float*)d_in[6];
  float* out = (float*)d_out;

  dim3 block(256);      // 8 waves/block -> fills a WGP twice
  dim3 grid(2048);      // 16384 waves, 8 grid-stride iterations each
  finn_burger2d_kernel<<<grid, block, 0, stream>>>(u, W1, W2, W3, Dc, BC, st, out);
}
